// RecurrentGRCX_41394894799291
// MI455X (gfx1250) — compile-verified, matrix operand order checked by reading, not run on hardware
//
#include <hip/hip_runtime.h>
#include <hip/hip_bf16.h>

// ---------------------------------------------------------------------------
// RecurrentGRC on MI455X (gfx1250).
// 8 independent row-groups (16 batch rows each); each group = 8 workgroups
// (column split) synced group-locally (cluster-barrier path + L2 ticket
// fallback). bf16 WMMA GEMMs; LDS staging via TDM (tensor_load_to_lds,
// TENSORcnt) for cat and global_load_async_to_lds_b128 (ASYNCcnt) for inter.
// ---------------------------------------------------------------------------

#define NB    128
#define SEQ   1024
#define HID   256
#define CH    1024
#define K1    512          // 2*HID
#define GROUPS 8           // independent row groups (16 rows each)
#define CSPL   8           // column-split workgroups per group
#define ROWS   16          // batch rows per group

// dynamic LDS layout (bytes)
#define LDS_CAT_OFF   0        // 16 x 512  bf16 = 16384
#define LDS_INTER_OFF 16384    // 16 x 1024 bf16 = 32768
#define LDS_CONT_OFF  49152    // 16 x 128  f32  =  8192
#define LDS_H_OFF     57344    // 16 x 32   f32  =  2048
#define LDS_BYTES     59392

typedef __attribute__((ext_vector_type(16))) __bf16 v16bf;
typedef __attribute__((ext_vector_type(8)))  float  v8f;
typedef __attribute__((ext_vector_type(4)))  unsigned int u32x4;
typedef __attribute__((ext_vector_type(8)))  int          i32x8;
typedef __attribute__((ext_vector_type(4)))  int          i32x4;

union FragU { uint4 u[2]; v16bf v; };

__device__ __forceinline__ unsigned short f2bf(float f) {
    unsigned int u = __float_as_uint(f);
    unsigned int r = u + 0x7FFFu + ((u >> 16) & 1u);   // round-to-nearest-even
    return (unsigned short)(r >> 16);
}

__device__ __forceinline__ float gelu_f(float v) {
    const float c = 0.7978845608028654f;
    float t = tanhf(c * (v + 0.044715f * v * v * v));
    return 0.5f * v * (1.0f + t);
}

__device__ __forceinline__ float sigmoid_f(float v) {
    return 1.0f / (1.0f + __expf(-v));
}

// --- CDNA5 async global->LDS copy (ASYNCcnt) -------------------------------
__device__ __forceinline__ void async_g2l_b128(unsigned lds_byte_off,
                                               const void* gsrc) {
    unsigned long long ga = (unsigned long long)(uintptr_t)gsrc;
    asm volatile("global_load_async_to_lds_b128 %0, %1, off"
                 :: "v"(lds_byte_off), "v"(ga) : "memory");
}
__device__ __forceinline__ void async_wait0() {
    asm volatile("s_wait_asynccnt 0x0" ::: "memory");
}

// --- CDNA5 Tensor Data Mover: 1D row of 2-byte elements -> LDS -------------
// D# per ISA ch.8: group0 {count=1, lds_addr, global_addr[56:0], type=2},
// group1 {data_size=1(2B), tensor_dim0=tile_dim0=nelem, tensor_dim1=tile_dim1=1,
// tensor_dim0_stride=nelem}; groups 2/3 zero (<=2D tensor).
// amdgpu-toolchain (clang-23) 6-arg builtin form.
__device__ __forceinline__ void tdm_copy_2b(unsigned lds_byte_off,
                                            const void* gsrc,
                                            unsigned nelem2b) {
    unsigned long long ga = (unsigned long long)(uintptr_t)gsrc;
    u32x4 g0;
    g0[0] = 1u;                                          // count=1 (user mode)
    g0[1] = lds_byte_off;                                // lds_addr
    g0[2] = (unsigned)(ga & 0xFFFFFFFFu);                // global_addr[31:0]
    g0[3] = (unsigned)((ga >> 32) & 0x01FFFFFFu)         // global_addr[56:32]
            | (2u << 30);                                // type=2 ("image")
    i32x8 g1;
    g1[0] = 0x00010000;                                  // data_size=1 -> 2B
    g1[1] = (int)((nelem2b & 0xFFFFu) << 16);            // tensor_dim0[15:0]
    g1[2] = (int)((nelem2b >> 16) & 0xFFFFu) | (1 << 16);// dim0 hi | tensor_dim1=1
    g1[3] = (int)((nelem2b & 0xFFFFu) << 16);            // tile_dim0
    g1[4] = 1;                                           // tile_dim1=1
    g1[5] = (int)nelem2b;                                // tensor_dim0_stride lo
    g1[6] = 0;
    g1[7] = 0;
    i32x4 gz;  gz[0] = 0; gz[1] = 0; gz[2] = 0; gz[3] = 0;
    i32x8 gz8; gz8[0] = 0; gz8[1] = 0; gz8[2] = 0; gz8[3] = 0;
    gz8[4] = 0; gz8[5] = 0; gz8[6] = 0; gz8[7] = 0;
    __builtin_amdgcn_tensor_load_to_lds(g0, g1, gz, gz, gz8, 0);
}

// --- group-local barrier: cluster path + 8-member L2 ticket fallback -------
__device__ __forceinline__ void group_barrier(int* gbar, int use_cluster) {
    __syncthreads();
    if (use_cluster) {
        __threadfence();
        __builtin_amdgcn_s_cluster_barrier();   // s_barrier_signal/wait -3
    } else if (threadIdx.x == 0) {
        volatile int* vb = (volatile int*)gbar;
        int gen = vb[1];
        __threadfence();
        int t = atomicAdd(&gbar[0], 1);
        if (t == CSPL - 1) {
            gbar[0] = 0;
            __threadfence();
            atomicAdd(&gbar[1], 1);
        } else {
            while (vb[1] == gen) __builtin_amdgcn_s_sleep(1);
        }
    }
    __syncthreads();
}

// ---------------------------------------------------------------------------
// Weight prep: f32 row-major W[K x Ncols] -> bf16 WMMA B-fragment order.
// ---------------------------------------------------------------------------
__global__ void swizzle_weight_kernel(const float* __restrict__ W,
                                      unsigned short* __restrict__ dst,
                                      int K, int Ncols, int gate_interleave) {
    int idx = blockIdx.x * blockDim.x + threadIdx.x;
    if (idx >= K * Ncols) return;
    int e    = idx & 15;
    int lane = (idx >> 4) & 31;
    int tile = idx >> 9;
    int nt_count = Ncols >> 4;
    int nt = tile % nt_count;
    int kt = tile / nt_count;
    int k = kt * 32 + ((e < 8) ? e : e + 8) + ((lane >= 16) ? 8 : 0);
    int n = nt * 16 + (lane & 15);
    int nsrc = gate_interleave ? ((n & 3) * (Ncols >> 2) + (n >> 2)) : n;
    dst[idx] = f2bf(W[k * Ncols + nsrc]);
}

// ---------------------------------------------------------------------------
// Scan kernel: grid = GROUPS*CSPL = 64 blocks x 256 threads (8 waves).
// Block (g,c): rows [16g,16g+16), permuted output cols [128c, 128c+128).
// ---------------------------------------------------------------------------
__global__ void __launch_bounds__(256, 1)
grc_scan_kernel(const float* __restrict__ x,        // (N,S,H)
                const float* __restrict__ maskp,    // (N,S)
                const float* __restrict__ b1,       // (CH)
                const float* __restrict__ b2,       // (CH)
                const float* __restrict__ gamma,    // (H)
                const float* __restrict__ beta,     // (H)
                const unsigned short* __restrict__ w1s,
                const unsigned short* __restrict__ w2s,   // gate-interleaved
                unsigned short* __restrict__ cat_ws,      // GROUPS x 16x512 bf16
                unsigned short* __restrict__ inter_ws,    // GROUPS x 16x1024 bf16
                float* __restrict__ sums_ws,              // GROUPS x 64 f32
                int* __restrict__ bars,                   // GROUPS x 64 int
                float* __restrict__ fseq,                 // (N,S,H)
                float* __restrict__ hfinal,               // (N,H)
                int use_cluster) {
    extern __shared__ char smem[];
    unsigned short* lds_cat   = (unsigned short*)(smem + LDS_CAT_OFF);
    unsigned short* lds_inter = (unsigned short*)(smem + LDS_INTER_OFF);
    float*          lds_cont  = (float*)(smem + LDS_CONT_OFF);
    float*          lds_h     = (float*)(smem + LDS_H_OFF);

    const int g = blockIdx.x >> 3;
    const int c = blockIdx.x & 7;
    const int tid   = threadIdx.x;
    const int wave  = tid >> 5;
    const int lane  = tid & 31;
    const int lhalf = lane & 15;
    const int koff  = (lane >= 16) ? 8 : 0;

    unsigned short* catg   = cat_ws   + (size_t)g * ROWS * K1;
    unsigned short* interg = inter_ws + (size_t)g * ROWS * CH;
    float* sums = sums_ws + g * 64;           // [2][16][2]
    int*   gbar = bars    + g * 64;

    const int nt  = c * 8 + wave;             // this wave's 16-col tile (0..63)
    const int n0  = nt * 16;
    const int jp  = lane;                     // update phase: column-in-block

    // ---- init: h=0, cat = [0 | bf16(x[:,0,:])], sums=0 --------------------
    for (int e = tid; e < ROWS * 32; e += 256) {
        int row = e >> 5, j = e & 31;
        lds_h[row * 32 + j] = 0.0f;
        int jg = c * 32 + j;
        catg[row * K1 + jg] = 0;
        catg[row * K1 + HID + jg] =
            f2bf(x[((size_t)(g * ROWS + row) * SEQ + 0) * HID + jg]);
    }
    if (c == 0 && tid < 64)
        __hip_atomic_store(&sums[tid], 0.0f,
                           __ATOMIC_RELAXED, __HIP_MEMORY_SCOPE_AGENT);
    group_barrier(gbar, use_cluster);

    for (int t = 0; t < SEQ; ++t) {
        const int buf = t & 1;

        // ---- stage cat (16KB) into LDS via TDM ----------------------------
        if (wave == 0) {
            tdm_copy_2b(LDS_CAT_OFF, catg, ROWS * K1);
            __builtin_amdgcn_s_wait_tensorcnt(0);
        }
        __syncthreads();

        // ---- GEMM1: inter[:, n0:n0+16) = gelu(cat @ W1 + b1) --------------
        {
            v8f acc = {};
            float bias = b1[n0 + lhalf];
            #pragma unroll 4
            for (int kt = 0; kt < K1 / 32; ++kt) {
                const unsigned short* arow =
                    lds_cat + lhalf * K1 + kt * 32 + koff;
                FragU a;
                a.u[0] = *(const uint4*)(arow);
                a.u[1] = *(const uint4*)(arow + 16);
                const v16bf b = *(const v16bf*)(
                    w1s + ((size_t)(kt * (CH / 16) + nt) * 32 + lane) * 16);
                if (kt + 1 < K1 / 32)
                    __builtin_prefetch(
                        w1s + ((size_t)((kt + 1) * (CH / 16) + nt) * 32 + lane) * 16,
                        0, 3);
                acc = __builtin_amdgcn_wmma_f32_16x16x32_bf16(
                    false, a.v, false, b, (short)0, acc, false, false);
            }
            #pragma unroll
            for (int r = 0; r < 8; ++r) {
                int mrow = r + koff;
                float v = acc[r] + bias;
                interg[mrow * CH + n0 + lhalf] = f2bf(gelu_f(v));
            }
        }
        group_barrier(gbar, use_cluster);      // inter visible group-wide

        // ---- stage inter (32KB) into LDS via async copy -------------------
        #pragma unroll
        for (int q = 0; q < 8; ++q) {
            int idx = tid + q * 256;                       // 2048 x 16B
            async_g2l_b128(LDS_INTER_OFF + idx * 16, interg + idx * 8);
        }
        async_wait0();
        __syncthreads();

        // ---- GEMM2: permuted contents[:, n0:n0+16) -> LDS -----------------
        {
            v8f acc = {};
            int np = n0 + lhalf;                           // permuted column
            float bias = b2[((np & 3) << 8) + (np >> 2)];
            #pragma unroll 4
            for (int kt = 0; kt < CH / 32; ++kt) {
                const unsigned short* arow =
                    lds_inter + lhalf * CH + kt * 32 + koff;
                FragU a;
                a.u[0] = *(const uint4*)(arow);
                a.u[1] = *(const uint4*)(arow + 16);
                const v16bf b = *(const v16bf*)(
                    w2s + ((size_t)(kt * (CH / 16) + nt) * 32 + lane) * 16);
                if (kt + 1 < CH / 32)
                    __builtin_prefetch(
                        w2s + ((size_t)((kt + 1) * (CH / 16) + nt) * 32 + lane) * 16,
                        0, 3);
                acc = __builtin_amdgcn_wmma_f32_16x16x32_bf16(
                    false, a.v, false, b, (short)0, acc, false, false);
            }
            #pragma unroll
            for (int r = 0; r < 8; ++r) {
                int mrow = r + koff;
                lds_cont[mrow * 128 + wave * 16 + lhalf] = acc[r] + bias;
            }
        }
        __syncthreads();

        // ---- gates + LN partial sums (rows: wave, wave+8; col jp) ---------
        float vals[2], holds[2];
        #pragma unroll
        for (int pidx = 0; pidx < 2; ++pidx) {
            int row = wave + pidx * 8;
            float f1c = sigmoid_f(lds_cont[row * 128 + jp * 4 + 0]);
            float f2c = sigmoid_f(lds_cont[row * 128 + jp * 4 + 1]);
            float ic  = sigmoid_f(lds_cont[row * 128 + jp * 4 + 2]);
            float pc  =           lds_cont[row * 128 + jp * 4 + 3];
            int jg = c * 32 + jp;
            float xv = x[((size_t)(g * ROWS + row) * SEQ + t) * HID + jg];
            float hv = lds_h[row * 32 + jp];
            holds[pidx] = hv;
            float v = f1c * hv + f2c * xv + ic * pc;
            vals[pidx] = v;
            float s = v, s2 = v * v;
            #pragma unroll
            for (int off = 16; off > 0; off >>= 1) {
                s  += __shfl_xor(s,  off, 32);
                s2 += __shfl_xor(s2, off, 32);
            }
            if (lane == 0) {
                atomicAdd(&sums[buf * 32 + row * 2 + 0], s);
                atomicAdd(&sums[buf * 32 + row * 2 + 1], s2);
            }
        }
        group_barrier(gbar, use_cluster);      // sums complete group-wide

        // ---- LN finalize + masked h update --------------------------------
        #pragma unroll
        for (int pidx = 0; pidx < 2; ++pidx) {
            int row = wave + pidx * 8;
            float S  = __hip_atomic_load(&sums[buf * 32 + row * 2 + 0],
                                         __ATOMIC_RELAXED, __HIP_MEMORY_SCOPE_AGENT);
            float S2 = __hip_atomic_load(&sums[buf * 32 + row * 2 + 1],
                                         __ATOMIC_RELAXED, __HIP_MEMORY_SCOPE_AGENT);
            float mean = S * (1.0f / HID);
            float var  = S2 * (1.0f / HID) - mean * mean;
            float rstd = rsqrtf(var + 1e-5f);
            int nrow = g * ROWS + row;
            int jg = c * 32 + jp;
            float mt = maskp[(size_t)nrow * SEQ + t];
            float ln = (vals[pidx] - mean) * rstd * gamma[jg] + beta[jg];
            float hn = mt * ln + (1.0f - mt) * holds[pidx];
            lds_h[row * 32 + jp] = hn;
            catg[row * K1 + jg] = f2bf(hn);
            fseq[((size_t)nrow * SEQ + t) * HID + jg] = hn;
            if (t + 1 < SEQ)
                catg[row * K1 + HID + jg] =
                    f2bf(x[((size_t)nrow * SEQ + t + 1) * HID + jg]);
            if (t == SEQ - 1)
                hfinal[nrow * HID + jg] = hn;
        }
        if (c == 0 && tid < 32)                // re-zero other sum buffer
            __hip_atomic_store(&sums[(buf ^ 1) * 32 + tid], 0.0f,
                               __ATOMIC_RELAXED, __HIP_MEMORY_SCOPE_AGENT);
        group_barrier(gbar, use_cluster);      // cat visible for next step
    }
}

// ---------------------------------------------------------------------------
extern "C" void kernel_launch(void* const* d_in, const int* in_sizes, int n_in,
                              void* d_out, int out_size, void* d_ws, size_t ws_size,
                              hipStream_t stream) {
    const float* x     = (const float*)d_in[0];
    const float* maskp = (const float*)d_in[1];
    const float* W1    = (const float*)d_in[2];
    const float* b1    = (const float*)d_in[3];
    const float* W2    = (const float*)d_in[4];
    const float* b2    = (const float*)d_in[5];
    const float* gamma = (const float*)d_in[6];
    const float* beta  = (const float*)d_in[7];

    float* out    = (float*)d_out;
    float* fseq   = out;
    float* hfinal = out + (size_t)NB * SEQ * HID;

    char* p = (char*)d_ws;
    unsigned short* w1s      = (unsigned short*)p; p += (size_t)K1 * CH * 2;        // 1 MB
    unsigned short* w2s      = (unsigned short*)p; p += (size_t)CH * CH * 2;        // 2 MB
    unsigned short* cat_ws   = (unsigned short*)p; p += (size_t)GROUPS * ROWS * K1 * 2;  // 128 KB
    unsigned short* inter_ws = (unsigned short*)p; p += (size_t)GROUPS * ROWS * CH * 2;  // 256 KB
    float* sums_ws           = (float*)p;          p += (size_t)GROUPS * 64 * 4;    // 2 KB
    int* bars                = (int*)p;            p += (size_t)GROUPS * 64 * 4;    // 2 KB

    (void)hipMemsetAsync(bars, 0, GROUPS * 64 * sizeof(int), stream);

    swizzle_weight_kernel<<<(K1 * CH + 255) / 256, 256, 0, stream>>>(
        W1, w1s, K1, CH, 0);
    swizzle_weight_kernel<<<(CH * CH + 255) / 256, 256, 0, stream>>>(
        W2, w2s, CH, CH, 1);

    grc_scan_kernel<<<GROUPS * CSPL, 256, LDS_BYTES, stream>>>(
        x, maskp, b1, b2, gamma, beta, w1s, w2s,
        cat_ws, inter_ws, sums_ws, bars, fseq, hfinal, /*use_cluster=*/0);
}
